// EntityEncoder_60670708023537
// MI455X (gfx1250) — matrix operand order repaired and checked by reading, not compiled, not run on hardware
//
#include <hip/hip_runtime.h>
#include <hip/hip_bf16.h>

typedef __bf16 bf16_t;
typedef __attribute__((ext_vector_type(16))) __bf16 v16bf;
typedef __attribute__((ext_vector_type(8)))  float v8f;

#define N_NODES 50000
#define N_EDGES 400000
#define DIM     128
#define HEADS   4

// ---------------------------------------------------------------------------
// Ordered-int trick float atomic max (works for mixed signs; init to -3e38)
// ---------------------------------------------------------------------------
static __device__ __forceinline__ void atomicMaxF(float* addr, float value) {
  if (value >= 0.f)
    atomicMax((int*)addr, __float_as_int(value));
  else
    atomicMin((unsigned int*)addr, __float_as_uint(value));
}

// ---------------------------------------------------------------------------
// Elementwise helpers
// ---------------------------------------------------------------------------
__global__ void k_f32_to_bf16(const float* __restrict__ in,
                              bf16_t* __restrict__ out, int n) {
  int i = blockIdx.x * blockDim.x + threadIdx.x;
  if (i < n) out[i] = (bf16_t)in[i];
}

__global__ void k_fill(float* __restrict__ p, float v, long long n) {
  long long i = (long long)blockIdx.x * blockDim.x + threadIdx.x;
  if (i < n) p[i] = v;
}

// ---------------------------------------------------------------------------
// WMMA GEMM:  C[M,N] = A[M,K] * B[N,K]^T   (A,B bf16; C f32)
// One wave computes one 16x16 tile of C; K multiple of 32.
// A fragment (16x32 bf16, contiguous k, not ISA table order): lane m=lane&15, half=lane>>4;
//   regs 0..7 hold K = half*16 + {0..15} (two bf16 per 32-bit reg)
// B fragment (32x16 bf16): lane n=lane&15, half=lane>>4;
//   regs 0..7 hold K = half*16 + {0..15} (two bf16 per 32-bit reg)
// C/D (16x16 f32): lane n=lane&15; reg r -> row m = r + 8*half
// ---------------------------------------------------------------------------
__global__ void __launch_bounds__(256)
k_gemm_bf16_wmma(const bf16_t* __restrict__ A, const bf16_t* __restrict__ B,
                 float* __restrict__ C, int M, int N, int K) {
  const int wave = threadIdx.x >> 5;
  const int lane = threadIdx.x & 31;
  const int tilesN = N >> 4;
  const int tile = blockIdx.x * 8 + wave;
  const int totalTiles = (M >> 4) * tilesN;
  if (tile >= totalTiles) return;
  const int tm = (tile / tilesN) << 4;
  const int tn = (tile % tilesN) << 4;

  const int mn   = lane & 15;   // row for A-frag, col for B/C-frag
  const int half = lane >> 4;
  const int akb  = half * 16;
  const int bkb  = half * 16;
  const bf16_t* Arow = A + (size_t)(tm + mn) * K;
  const bf16_t* Brow = B + (size_t)(tn + mn) * K;

  v8f c = {0.f, 0.f, 0.f, 0.f, 0.f, 0.f, 0.f, 0.f};

  for (int k0 = 0; k0 < K; k0 += 32) {
    union { v16bf v; unsigned u[8]; } a, b;
#pragma unroll
    for (int i = 0; i < 4; ++i) {
      a.u[i]     = *(const unsigned*)(Arow + k0 + akb + 2 * i);
      a.u[4 + i] = *(const unsigned*)(Arow + k0 + akb + 8 + 2 * i);
    }
#pragma unroll
    for (int i = 0; i < 8; ++i) {
      b.u[i] = *(const unsigned*)(Brow + k0 + bkb + 2 * i);
    }
    c = __builtin_amdgcn_wmma_f32_16x16x32_bf16(false, a.v, false, b.v,
                                                (short)0, c, false, false);
  }

  float* Crow = C + (size_t)tm * N + tn;
#pragma unroll
  for (int r = 0; r < 8; ++r) {
    Crow[(size_t)(r + half * 8) * N + mn] = c[r];
  }
}

// ---------------------------------------------------------------------------
// Edge attention scores: one thread per (edge, head).
//   alpha[e,h] = sum_c leaky_relu(xd[dst,h,c] + xs[src,h,c] + ew[e]*wedge[h,c]) * att[h,c]
//   also segment-max into amax[dst,h].
// C is 32 (layer 0) or 128 (layer 1); rows are H*C floats, 16B aligned.
// ---------------------------------------------------------------------------
__global__ void k_edge_scores(const int* __restrict__ src, const int* __restrict__ dst,
                              const float* __restrict__ ew,
                              const float* __restrict__ xs, const float* __restrict__ xd,
                              const float* __restrict__ att, const float* __restrict__ wedge,
                              float* __restrict__ alpha, float* __restrict__ amax,
                              int E, int C) {
  int t = blockIdx.x * blockDim.x + threadIdx.x;
  if (t >= E * HEADS) return;
  const int e = t >> 2;
  const int h = t & 3;
  const int s = src[e];
  const int d = dst[e];
  const float w = ew[e];
  const int Drow = HEADS * C;
  const float4* ps = (const float4*)(xs + (size_t)s * Drow + h * C);
  const float4* pd = (const float4*)(xd + (size_t)d * Drow + h * C);
  const float4* pa = (const float4*)(att + h * C);
  const float4* pw = (const float4*)(wedge + h * C);
  float sum = 0.f;
  for (int c4 = 0; c4 < (C >> 2); ++c4) {
    float4 vs = ps[c4], vd = pd[c4], va = pa[c4], vw = pw[c4];
    float v0 = vs.x + vd.x + w * vw.x;
    float v1 = vs.y + vd.y + w * vw.y;
    float v2 = vs.z + vd.z + w * vw.z;
    float v3 = vs.w + vd.w + w * vw.w;
    v0 = (v0 > 0.f) ? v0 : 0.2f * v0;
    v1 = (v1 > 0.f) ? v1 : 0.2f * v1;
    v2 = (v2 > 0.f) ? v2 : 0.2f * v2;
    v3 = (v3 > 0.f) ? v3 : 0.2f * v3;
    sum += v0 * va.x + v1 * va.y + v2 * va.z + v3 * va.w;
  }
  alpha[t] = sum;
  atomicMaxF(&amax[(size_t)d * HEADS + h], sum);
}

// ---------------------------------------------------------------------------
// Segment softmax numerator: alpha <- exp(alpha - amax[dst]); denom += alpha
// ---------------------------------------------------------------------------
__global__ void k_edge_exp(const int* __restrict__ dst,
                           float* __restrict__ alpha,
                           const float* __restrict__ amax,
                           float* __restrict__ denom, int E) {
  int t = blockIdx.x * blockDim.x + threadIdx.x;
  if (t >= E * HEADS) return;
  const int e = t >> 2;
  const int h = t & 3;
  const int d = dst[e];
  const float v = __expf(alpha[t] - amax[(size_t)d * HEADS + h]);
  alpha[t] = v;
  atomicAdd(&denom[(size_t)d * HEADS + h], v);
}

// ---------------------------------------------------------------------------
// Weighted scatter: one thread per (edge, channel).
//   acc[dst, ch] += xs[src, ch] * alpha[e,h] / (denom[dst,h] + 1e-16)
// dshift = log2(H*C): 7 (layer 0) or 9 (layer 1); cshift = log2(C).
// ---------------------------------------------------------------------------
__global__ void k_edge_scatter(const int* __restrict__ src, const int* __restrict__ dst,
                               const float* __restrict__ alpha,
                               const float* __restrict__ denom,
                               const float* __restrict__ xs,
                               float* __restrict__ acc,
                               long long total, int dshift, int cshift) {
  long long t = (long long)blockIdx.x * blockDim.x + threadIdx.x;
  if (t >= total) return;
  const int e  = (int)(t >> dshift);
  const int ch = (int)(t & ((1 << dshift) - 1));
  const int h  = ch >> cshift;
  const int s  = src[e];
  const int d  = dst[e];
  const float coef = alpha[(size_t)e * HEADS + h] /
                     (denom[(size_t)d * HEADS + h] + 1e-16f);
  atomicAdd(&acc[((size_t)d << dshift) + ch],
            xs[((size_t)s << dshift) + ch] * coef);
}

// ---------------------------------------------------------------------------
// Layer-0 finalize: x1 = elu(acc + bias), emitted directly as bf16 for layer-1 GEMM
// ---------------------------------------------------------------------------
__global__ void k_finalize0(const float* __restrict__ acc,
                            const float* __restrict__ bias,
                            bf16_t* __restrict__ x1b, int n) {
  int i = blockIdx.x * blockDim.x + threadIdx.x;
  if (i >= n) return;
  float v = acc[i] + bias[i & (DIM - 1)];
  v = (v > 0.f) ? v : (__expf(v) - 1.f);
  x1b[i] = (bf16_t)v;
}

// ---------------------------------------------------------------------------
// Layer-1 finalize: out[n,d] = mean_h acc[n, h*128 + d] + bias[d]
// ---------------------------------------------------------------------------
__global__ void k_finalize1(const float* __restrict__ acc,
                            const float* __restrict__ bias,
                            float* __restrict__ out, int n) {
  int i = blockIdx.x * blockDim.x + threadIdx.x;
  if (i >= n) return;
  const int node = i >> 7;
  const int d = i & 127;
  const float* p = acc + (size_t)node * (HEADS * DIM) + d;
  out[i] = 0.25f * (p[0] + p[DIM] + p[2 * DIM] + p[3 * DIM]) + bias[d];
}

// ---------------------------------------------------------------------------
// Host launcher
// ---------------------------------------------------------------------------
extern "C" void kernel_launch(void* const* d_in, const int* in_sizes, int n_in,
                              void* d_out, int out_size, void* d_ws, size_t ws_size,
                              hipStream_t stream) {
  const int*   ei       = (const int*)d_in[0];        // [2, E]
  const float* ew       = (const float*)d_in[1];      // [E, 1]
  const float* emb      = (const float*)d_in[2];      // [N, 128]
  const float* l0_wsrc  = (const float*)d_in[3];      // [128, 128]
  const float* l0_wdst  = (const float*)d_in[4];      // [128, 128]
  const float* l0_att   = (const float*)d_in[5];      // [1, 4, 32]
  const float* l0_wedge = (const float*)d_in[6];      // [128, 1]
  const float* l0_bias  = (const float*)d_in[7];      // [128]
  const float* l1_wsrc  = (const float*)d_in[8];      // [512, 128]
  const float* l1_wdst  = (const float*)d_in[9];      // [512, 128]
  const float* l1_att   = (const float*)d_in[10];     // [1, 4, 128]
  const float* l1_wedge = (const float*)d_in[11];     // [512, 1]
  const float* l1_bias  = (const float*)d_in[12];     // [128]
  float* out = (float*)d_out;

  const int N = N_NODES, E = N_EDGES;
  const int* src = ei;
  const int* dst = ei + E;

  // --- workspace layout (256B aligned slices) ---
  char* ws = (char*)d_ws;
  size_t off = 0;
  auto alloc = [&](size_t bytes) -> void* {
    off = (off + 255) & ~(size_t)255;
    void* p = ws + off;
    off += bytes;
    return p;
  };
  bf16_t* x0b   = (bf16_t*)alloc((size_t)N * DIM * 2);
  bf16_t* w0s_b = (bf16_t*)alloc((size_t)DIM * DIM * 2);
  bf16_t* w0d_b = (bf16_t*)alloc((size_t)DIM * DIM * 2);
  bf16_t* w1s_b = (bf16_t*)alloc((size_t)4 * DIM * DIM * 2);
  bf16_t* w1d_b = (bf16_t*)alloc((size_t)4 * DIM * DIM * 2);
  float*  xs0   = (float*)alloc((size_t)N * DIM * 4);
  float*  xd0   = (float*)alloc((size_t)N * DIM * 4);
  float*  alpha = (float*)alloc((size_t)E * HEADS * 4);
  float*  amax  = (float*)alloc((size_t)N * HEADS * 4);
  float*  den   = (float*)alloc((size_t)N * HEADS * 4);
  float*  acc0  = (float*)alloc((size_t)N * DIM * 4);
  bf16_t* x1b   = (bf16_t*)alloc((size_t)N * DIM * 2);
  float*  xs1   = (float*)alloc((size_t)N * 4 * DIM * 4);
  float*  xd1   = (float*)alloc((size_t)N * 4 * DIM * 4);
  float*  acc1  = (float*)alloc((size_t)N * 4 * DIM * 4);
  (void)ws_size; (void)n_in; (void)in_sizes; (void)out_size;

  const int B = 256;
  auto grid = [](long long n, int b) { return (unsigned)((n + b - 1) / b); };

  // ---- convert inputs to bf16 ----
  k_f32_to_bf16<<<grid((long long)N * DIM, B), B, 0, stream>>>(emb, x0b, N * DIM);
  k_f32_to_bf16<<<grid(DIM * DIM, B), B, 0, stream>>>(l0_wsrc, w0s_b, DIM * DIM);
  k_f32_to_bf16<<<grid(DIM * DIM, B), B, 0, stream>>>(l0_wdst, w0d_b, DIM * DIM);
  k_f32_to_bf16<<<grid(4 * DIM * DIM, B), B, 0, stream>>>(l1_wsrc, w1s_b, 4 * DIM * DIM);
  k_f32_to_bf16<<<grid(4 * DIM * DIM, B), B, 0, stream>>>(l1_wdst, w1d_b, 4 * DIM * DIM);

  // ---- layer 0 projections via WMMA ----
  {
    const int tiles = (N / 16) * (DIM / 16);           // 25000
    k_gemm_bf16_wmma<<<(tiles + 7) / 8, 256, 0, stream>>>(x0b, w0s_b, xs0, N, DIM, DIM);
    k_gemm_bf16_wmma<<<(tiles + 7) / 8, 256, 0, stream>>>(x0b, w0d_b, xd0, N, DIM, DIM);
  }

  // ---- layer 0 edge phase ----
  k_fill<<<grid((long long)N * HEADS, B), B, 0, stream>>>(amax, -3e38f, (long long)N * HEADS);
  k_fill<<<grid((long long)N * HEADS, B), B, 0, stream>>>(den, 0.f, (long long)N * HEADS);
  k_fill<<<grid((long long)N * DIM, B), B, 0, stream>>>(acc0, 0.f, (long long)N * DIM);
  k_edge_scores<<<grid((long long)E * HEADS, B), B, 0, stream>>>(
      src, dst, ew, xs0, xd0, l0_att, l0_wedge, alpha, amax, E, DIM / HEADS);
  k_edge_exp<<<grid((long long)E * HEADS, B), B, 0, stream>>>(dst, alpha, amax, den, E);
  k_edge_scatter<<<grid((long long)E * DIM, B), B, 0, stream>>>(
      src, dst, alpha, den, xs0, acc0, (long long)E * DIM, 7, 5);
  k_finalize0<<<grid((long long)N * DIM, B), B, 0, stream>>>(acc0, l0_bias, x1b, N * DIM);

  // ---- layer 1 projections via WMMA ----
  {
    const int tiles = (N / 16) * (4 * DIM / 16);       // 100000
    k_gemm_bf16_wmma<<<(tiles + 7) / 8, 256, 0, stream>>>(x1b, w1s_b, xs1, N, 4 * DIM, DIM);
    k_gemm_bf16_wmma<<<(tiles + 7) / 8, 256, 0, stream>>>(x1b, w1d_b, xd1, N, 4 * DIM, DIM);
  }

  // ---- layer 1 edge phase ----
  k_fill<<<grid((long long)N * HEADS, B), B, 0, stream>>>(amax, -3e38f, (long long)N * HEADS);
  k_fill<<<grid((long long)N * HEADS, B), B, 0, stream>>>(den, 0.f, (long long)N * HEADS);
  k_fill<<<grid((long long)N * 4 * DIM, B), B, 0, stream>>>(acc1, 0.f, (long long)N * 4 * DIM);
  k_edge_scores<<<grid((long long)E * HEADS, B), B, 0, stream>>>(
      src, dst, ew, xs1, xd1, l1_att, l1_wedge, alpha, amax, E, DIM);
  k_edge_exp<<<grid((long long)E * HEADS, B), B, 0, stream>>>(dst, alpha, amax, den, E);
  k_edge_scatter<<<grid((long long)E * 4 * DIM, B), B, 0, stream>>>(
      src, dst, alpha, den, xs1, acc1, (long long)E * 4 * DIM, 9, 7);
  k_finalize1<<<grid((long long)N * DIM, B), B, 0, stream>>>(acc1, l1_bias, out, N * DIM);
}
